// GridNetBlock_43739946943139
// MI455X (gfx1250) — matrix operand order repaired and checked by reading, not compiled, not run on hardware
//
#include <hip/hip_runtime.h>
#include <math.h>

// ---------------------------------------------------------------------------
// Types / helpers
// ---------------------------------------------------------------------------
typedef __attribute__((ext_vector_type(16))) __bf16 v16bf;
typedef __attribute__((ext_vector_type(8)))  float  v8f;
// TDM descriptor operand types (6-arg clang-23 builtin form)
typedef __attribute__((ext_vector_type(4))) unsigned tdm_v4u;
typedef __attribute__((ext_vector_type(8))) int      tdm_v8i;
typedef __attribute__((ext_vector_type(4))) int      tdm_v4i;

__device__ __forceinline__ __bf16 f2bf(float f) {
  unsigned u = __builtin_bit_cast(unsigned, f);
  unsigned r = u + 0x7FFFu + ((u >> 16) & 1u);     // round-to-nearest-even
  unsigned short h = (unsigned short)(r >> 16);
  return __builtin_bit_cast(__bf16, h);
}

union AFrag { v16bf v; uint4 q[2]; };

// A-matrix fragment (16x32 bf16), source row-major with row stride `rowStride`
__device__ __forceinline__ v16bf load_frag_a(const __bf16* row0, long rowStride,
                                             int kbase, int lane) {
  int m = lane & 15, hi = lane >> 4;
  const __bf16* r = row0 + (long)m * rowStride + kbase;
  AFrag f;
  f.q[0] = *(const uint4*)(r + 16 * hi);
  f.q[1] = *(const uint4*)(r + 16 * hi + 8);
  return f.v;
}

// B-matrix fragment (32x16 bf16) where B = X^T, X row-major (rows = B columns).
__device__ __forceinline__ v16bf load_frag_b_rowK(const __bf16* row0, long rowStride,
                                                  int kbase, int lane) {
  int n = lane & 15, hi = lane >> 4;
  const __bf16* r = row0 + (long)n * rowStride + kbase + 16 * hi;
  AFrag f;
  f.q[0] = *(const uint4*)(r);
  f.q[1] = *(const uint4*)(r + 8);
  return f.v;
}

// Pre-swizzled B fragment: 16 contiguous bf16 per lane.
__device__ __forceinline__ v16bf load_frag_b_sw(const __bf16* base, int lane) {
  const __bf16* r = base + lane * 16;
  AFrag f;
  f.q[0] = *(const uint4*)(r);
  f.q[1] = *(const uint4*)(r + 8);
  return f.v;
}

__device__ __forceinline__ v8f wmma_bf16(v16bf a, v16bf b, v8f c) {
  return __builtin_amdgcn_wmma_f32_16x16x32_bf16(false, a, false, b, (short)0, c,
                                                 false, false);
}

__device__ __forceinline__ float sigm(float x) { return 1.f / (1.f + __expf(-x)); }

// ---------------------------------------------------------------------------
// K1: LayerNorm over channel dim C of (B,C,T,F)
// ---------------------------------------------------------------------------
__global__ void k_ln_c(const float* __restrict__ x, const float* __restrict__ g,
                       const float* __restrict__ b, float* __restrict__ z,
                       int C, long TF, long total) {
  long idx = (long)blockIdx.x * blockDim.x + threadIdx.x;  // over B*T*F
  if (idx >= total) return;
  long bb = idx / TF, p = idx % TF;
  const float* xp = x + bb * C * TF + p;
  float mu = 0.f;
  for (int c = 0; c < C; ++c) mu += xp[(long)c * TF];
  mu /= (float)C;
  float m2 = 0.f;
  for (int c = 0; c < C; ++c) { float v = xp[(long)c * TF] - mu; m2 += v * v; }
  float inv = rsqrtf(m2 / (float)C + 1e-5f);
  float* zp = z + bb * C * TF + p;
  for (int c = 0; c < C; ++c)
    zp[(long)c * TF] = (xp[(long)c * TF] - mu) * inv * g[c] + b[c];
}

// ---------------------------------------------------------------------------
// K2: unfold (KS=4,HS=1) + bf16 convert.  U[(n*L+l)*192 + c*4+k]
// ---------------------------------------------------------------------------
__global__ void k_unfold(const float* __restrict__ z, __bf16* __restrict__ U,
                         int C, int T, int F, int L, long total, int mode) {
  long idx = (long)blockIdx.x * blockDim.x + threadIdx.x;
  if (idx >= total) return;
  int d = (int)(idx % 192);
  long r = idx / 192;
  int l = (int)(r % L);
  long n = r / L;
  int c = d >> 2, k = d & 3;
  float v;
  if (mode == 0) {
    long t = n % T, b = n / T;
    v = z[((b * C + c) * T + t) * F + (l + k)];
  } else {
    long q = n % F, b = n / F;
    v = z[((b * C + c) * T + (l + k)) * F + q];
  }
  U[idx] = f2bf(v);
}

// ---------------------------------------------------------------------------
// K3: pack a (768 x 192) weight into WMMA-B fragment order, bf16.
// ---------------------------------------------------------------------------
__global__ void k_pack_w(const float* __restrict__ W, __bf16* __restrict__ S) {
  int idx = blockIdx.x * blockDim.x + threadIdx.x;
  if (idx >= 768 * 192) return;
  int e = idx & 15, lane = (idx >> 4) & 31, frag = idx >> 9;
  int kt = frag % 6, nt = frag / 6;
  int n = lane & 15, hi = lane >> 4;
  int K = kt * 32 + 16 * hi + e;
  int N = nt * 16 + n;
  S[idx] = f2bf(W[N * 192 + K]);
}

// ---------------------------------------------------------------------------
// K4: persistent BiLSTM scan.  One WG = 16 sequences x one direction.
//     12 waves; wave w owns hidden tile j in [16w,16w+16) with 4 gate accums.
//     Whh (bf16, fragment order, 288KB) DMA'd into LDS by the Tensor Data
//     Mover (one wave issues tensor_load_to_lds, tracked on TENSORcnt);
//     h double-buffered in LDS.
// ---------------------------------------------------------------------------
#define WHH_ELEMS (192 * 768)
#define WHH_BYTES (WHH_ELEMS * 2)
#define WHH_QW    (WHH_BYTES / 8)   // 36864 8-byte elements

__global__ __launch_bounds__(384) void k_lstm_scan(
    const __bf16* __restrict__ U,     // (nSeqPad, L, 192)
    const __bf16* __restrict__ WihS,  // 2 dirs x 147456, fragment order
    const __bf16* __restrict__ WhhS,  // 2 dirs x 147456, fragment order
    const float*  __restrict__ bias,  // (2, 768)
    float* __restrict__ hs,           // (nSeqPad, L, 384)
    int L) {
  extern __shared__ char smem[];
  __bf16* whh  = (__bf16*)smem;                  // 147456 elems
  __bf16* hbuf = (__bf16*)(smem + WHH_BYTES);    // 2 * (16*192)

  const int dir = blockIdx.y;
  const __bf16* wih  = WihS + (long)dir * WHH_ELEMS;
  const __bf16* whhG = WhhS + (long)dir * WHH_ELEMS;
  const float*  bptr = bias + dir * 768;

  const int tid = threadIdx.x;

#if defined(__gfx1250__) && __has_builtin(__builtin_amdgcn_tensor_load_to_lds)
  // --- TDM bulk preload of Whh into LDS (ISA 08_async_tensor §7/§8) ---
  // 1-D D#: data_size=8B, tile_dim0 = tensor_dim0 = 36864 elems = 294912 B.
  if (tid < 32) {                      // one wave issues the DMA (EXEC ignored)
    unsigned lds_off = (unsigned)(size_t)whh;          // addr[31:0] = LDS byte addr
    unsigned long long ga = (unsigned long long)(size_t)whhG;
    tdm_v4u g0;
    g0[0] = 1u;                                        // count=1, user mode
    g0[1] = lds_off;                                   // lds_addr
    g0[2] = (unsigned)(ga & 0xFFFFFFFFu);              // global_addr[31:0]
    g0[3] = (unsigned)(((ga >> 32) & 0x1FFFFFFu) | (2u << 30)); // addr[56:32]|type=2
    tdm_v8i g1;
    g1[0] = (int)(3u << 16);                           // workgroup_mask=0, data_size=8B
    g1[1] = (int)((WHH_QW & 0xFFFFu) << 16);           // tensor_dim0[15:0]
    g1[2] = (int)((WHH_QW >> 16) & 0xFFFFu);           // tensor_dim0[31:16], tensor_dim1=0
    g1[3] = (int)((WHH_QW & 0xFFFFu) << 16);           // tile_dim0 = 36864
    g1[4] = 0;                                         // tile_dim1=0 (1-D), tile_dim2=0
    g1[5] = (int)WHH_QW;                               // tensor_dim0_stride[31:0]
    g1[6] = 0;
    g1[7] = 0;
    tdm_v4i g2 = {0, 0, 0, 0};
    tdm_v4i g3 = {0, 0, 0, 0};
    tdm_v8i g4 = {0, 0, 0, 0, 0, 0, 0, 0};             // unused trailing group
    __builtin_amdgcn_tensor_load_to_lds(g0, g1, g2, g3, g4, 0);
    __builtin_amdgcn_s_wait_tensorcnt(0);
  }
#else
  for (int i = tid; i < WHH_ELEMS / 8; i += 384)       // fallback: uint4 copy
    ((uint4*)whh)[i] = ((const uint4*)whhG)[i];
#endif
  for (int i = tid; i < 2 * 16 * 192; i += 384) hbuf[i] = f2bf(0.f);
#if defined(__gfx1250__)
  __builtin_amdgcn_s_cluster_barrier();   // NOP outside clusters (ISA 3.2)
#endif
  __syncthreads();

  const int wave = tid >> 5, lane = tid & 31;
  const int jt = wave;                      // hidden tile 0..11
  const int n = lane & 15, hi = lane >> 4;

  const float b0 = bptr[0 * 192 + jt * 16 + n];
  const float b1 = bptr[1 * 192 + jt * 16 + n];
  const float b2 = bptr[2 * 192 + jt * 16 + n];
  const float b3 = bptr[3 * 192 + jt * 16 + n];

  float cst[8];
#pragma unroll
  for (int r = 0; r < 8; ++r) cst[r] = 0.f;

  const long seqBase = (long)blockIdx.x * 16;
  const long uRowStride = (long)L * 192;
  int cur = 0;

  for (int s = 0; s < L; ++s) {
    const int t = dir ? (L - 1 - s) : s;
    v8f ai, af_, ag, ao;
#pragma unroll
    for (int r = 0; r < 8; ++r) { ai[r] = b0; af_[r] = b1; ag[r] = b2; ao[r] = b3; }

    const __bf16* urow0 = U + (seqBase * L + t) * 192;
    {  // prefetch next step's rows into cache hierarchy
      const int tn = dir ? (t > 0 ? t - 1 : 0) : (t + 1 < L ? t + 1 : t);
      __builtin_prefetch(U + (seqBase * L + tn) * 192 + (lane << 3), 0, 1);
    }
#pragma unroll
    for (int kt = 0; kt < 6; ++kt) {     // input projection, K = 192
      v16bf a = load_frag_a(urow0, uRowStride, kt * 32, lane);
      ai  = wmma_bf16(a, load_frag_b_sw(wih + (((0 * 12 + jt) * 6) + kt) * 512, lane), ai);
      af_ = wmma_bf16(a, load_frag_b_sw(wih + (((1 * 12 + jt) * 6) + kt) * 512, lane), af_);
      ag  = wmma_bf16(a, load_frag_b_sw(wih + (((2 * 12 + jt) * 6) + kt) * 512, lane), ag);
      ao  = wmma_bf16(a, load_frag_b_sw(wih + (((3 * 12 + jt) * 6) + kt) * 512, lane), ao);
    }
    const __bf16* hrow0 = hbuf + cur * (16 * 192);
#pragma unroll
    for (int kt = 0; kt < 6; ++kt) {     // recurrent projection, K = 192 (LDS)
      v16bf a = load_frag_a(hrow0, 192, kt * 32, lane);
      ai  = wmma_bf16(a, load_frag_b_sw(whh + (((0 * 12 + jt) * 6) + kt) * 512, lane), ai);
      af_ = wmma_bf16(a, load_frag_b_sw(whh + (((1 * 12 + jt) * 6) + kt) * 512, lane), af_);
      ag  = wmma_bf16(a, load_frag_b_sw(whh + (((2 * 12 + jt) * 6) + kt) * 512, lane), ag);
      ao  = wmma_bf16(a, load_frag_b_sw(whh + (((3 * 12 + jt) * 6) + kt) * 512, lane), ao);
    }

    __bf16* hnext = hbuf + (cur ^ 1) * (16 * 192);
#pragma unroll
    for (int r = 0; r < 8; ++r) {        // i/f/g/o co-resident per lane: no shuffles
      const int m = r + 8 * hi;
      float iv = sigm(ai[r]);
      float fv = sigm(af_[r]);
      float gv = tanhf(ag[r]);
      float ov = sigm(ao[r]);
      float cv = fv * cst[r] + iv * gv;
      cst[r] = cv;
      float hv = ov * tanhf(cv);
      hnext[m * 192 + jt * 16 + n] = f2bf(hv);
      hs[((seqBase + m) * (long)L + t) * 384 + dir * 192 + jt * 16 + n] = hv;
    }
    __syncthreads();
    cur ^= 1;
  }
}

// ---------------------------------------------------------------------------
// K5: ConvTranspose1d (KS=4,HS=1) + bias + residual.  TW layout (384,48,4).
// ---------------------------------------------------------------------------
__global__ void k_convt_res(const float* __restrict__ hs, const float* __restrict__ TW,
                            const float* __restrict__ Tb, const float* __restrict__ res,
                            float* __restrict__ out, int C, int T, int F, int L,
                            long total, int mode) {
  long idx = (long)blockIdx.x * blockDim.x + threadIdx.x;
  if (idx >= total) return;
  int f = (int)(idx % F);
  long r = idx / F;
  int t = (int)(r % T); r /= T;
  int c = (int)(r % C);
  long b = r / C;
  long nn; int q;
  if (mode == 0) { nn = b * T + t; q = f; }
  else           { nn = b * F + f; q = t; }
  float acc = Tb[c];
  for (int k = 0; k < 4; ++k) {
    int l = q - k;
    if (l < 0 || l >= L) continue;
    const float* hp = hs + (nn * L + l) * 384;
    const float* wp = TW + c * 4 + k;   // stride 192 per input channel
    float sum = 0.f;
    for (int i = 0; i < 384; ++i) sum += hp[i] * wp[i * 192];
    acc += sum;
  }
  out[idx] = acc + res[idx];
}

// ---------------------------------------------------------------------------
// K6: attention head projection + LeakyReLU + LN over (E,F) per (b,h,t).
// ---------------------------------------------------------------------------
__global__ void k_heads(const float* __restrict__ z, const float* __restrict__ W,
                        const float* __restrict__ bb, const float* __restrict__ alpha,
                        const float* __restrict__ g, const float* __restrict__ bg,
                        __bf16* __restrict__ out, int C, int T, int F, int H, int E,
                        int ldo, int mode) {
  const int bh_t = blockIdx.x;
  const int t = bh_t % T;
  const int bh = bh_t / T;
  const int h = bh % H;
  const long b = bh / H;
  const int EF = E * F;
  const float a = alpha[h];
  __shared__ float red[256];
  float vals[4];
  float lsum = 0.f;
#pragma unroll
  for (int ii = 0; ii < 4; ++ii) {
    int p = threadIdx.x + ii * 256;
    float v = 0.f;
    if (p < EF) {
      int e = p / F, f = p % F;
      const float* wp = W + (h * E + e) * C;
      const float* zb = z + b * C * T * F + (long)t * F + f;
      float s = bb[h * E + e];
      for (int c = 0; c < C; ++c) s += zb[(long)c * T * F] * wp[c];
      v = s >= 0.f ? s : a * s;
      lsum += v;
    }
    vals[ii] = v;
  }
  red[threadIdx.x] = lsum; __syncthreads();
  for (int o = 128; o > 0; o >>= 1) {
    if (threadIdx.x < o) red[threadIdx.x] += red[threadIdx.x + o];
    __syncthreads();
  }
  const float mu = red[0] / (float)EF; __syncthreads();
  float lvar = 0.f;
#pragma unroll
  for (int ii = 0; ii < 4; ++ii) {
    int p = threadIdx.x + ii * 256;
    if (p < EF) { float d = vals[ii] - mu; lvar += d * d; }
  }
  red[threadIdx.x] = lvar; __syncthreads();
  for (int o = 128; o > 0; o >>= 1) {
    if (threadIdx.x < o) red[threadIdx.x] += red[threadIdx.x + o];
    __syncthreads();
  }
  const float inv = rsqrtf(red[0] / (float)EF + 1e-5f);
#pragma unroll
  for (int ii = 0; ii < 4; ++ii) {
    int p = threadIdx.x + ii * 256;
    if (p < EF) {
      int e = p / F, f = p % F;
      float o_ = (vals[ii] - mu) * inv * g[(h * E + e) * F + f] + bg[(h * E + e) * F + f];
      if (mode == 0) out[((long)bh * T + t) * ldo + p] = f2bf(o_);
      else           out[((long)bh * 784 + p) * 512 + t] = f2bf(o_);
    }
  }
}

// ---------------------------------------------------------------------------
// K7a: S = (Qf @ Kf^T) * scale via WMMA.  Qf/Kf: (BH, T, lda=544) bf16.
// ---------------------------------------------------------------------------
__global__ __launch_bounds__(256) void k_scores(const __bf16* __restrict__ Qf,
                                                const __bf16* __restrict__ Kf,
                                                float* __restrict__ S, int T, int KT,
                                                int lda, float scale, int tilesN) {
  const int wave = threadIdx.x >> 5, lane = threadIdx.x & 31;
  const int gw = blockIdx.x * 8 + wave;
  const int tilesPerBH = tilesN * tilesN;
  const int bh = gw / tilesPerBH;
  const int tt = gw % tilesPerBH;
  const int mt = tt / tilesN, nt = tt % tilesN;
  const __bf16* qb = Qf + ((long)bh * T + mt * 16) * lda;
  const __bf16* kb = Kf + ((long)bh * T + nt * 16) * lda;
  v8f acc = {};
  for (int kt = 0; kt < KT; ++kt) {
    v16bf a = load_frag_a(qb, lda, kt * 32, lane);
    v16bf b = load_frag_b_rowK(kb, lda, kt * 32, lane);
    acc = wmma_bf16(a, b, acc);
  }
  const int n = lane & 15, hi = lane >> 4;
  float* sp = S + ((long)bh * T + mt * 16) * T + nt * 16;
#pragma unroll
  for (int r = 0; r < 8; ++r) sp[(long)(r + 8 * hi) * T + n] = acc[r] * scale;
}

// ---------------------------------------------------------------------------
// K7b: row softmax (rows = BH*T, width T), write bf16 P.
// ---------------------------------------------------------------------------
__global__ void k_softmax(const float* __restrict__ S, __bf16* __restrict__ P, int T) {
  const long row = blockIdx.x;
  const float* sp = S + row * T;
  __shared__ float red[256];
  const int tid = threadIdx.x;
  float m = -1e30f;
  for (int i = tid; i < T; i += 256) m = fmaxf(m, sp[i]);
  red[tid] = m; __syncthreads();
  for (int o = 128; o > 0; o >>= 1) {
    if (tid < o) red[tid] = fmaxf(red[tid], red[tid + o]);
    __syncthreads();
  }
  m = red[0]; __syncthreads();
  float s = 0.f;
  for (int i = tid; i < T; i += 256) s += __expf(sp[i] - m);
  red[tid] = s; __syncthreads();
  for (int o = 128; o > 0; o >>= 1) {
    if (tid < o) red[tid] += red[tid + o];
    __syncthreads();
  }
  const float inv = 1.f / red[0];
  for (int i = tid; i < T; i += 256) P[row * T + i] = f2bf(__expf(sp[i] - m) * inv);
}

// ---------------------------------------------------------------------------
// K7c: O = P @ Vf via WMMA.  P:(BH,T,T) bf16, VT:(BH,784,512) bf16, O:(BH,T,784).
// ---------------------------------------------------------------------------
__global__ __launch_bounds__(256) void k_attn_o(const __bf16* __restrict__ P,
                                                const __bf16* __restrict__ VT,
                                                float* __restrict__ O, int T, int NT) {
  const int wave = threadIdx.x >> 5, lane = threadIdx.x & 31;
  const int gw = blockIdx.x * 8 + wave;
  const int tilesPerBH = (T / 16) * NT;
  const int bh = gw / tilesPerBH;
  const int tt = gw % tilesPerBH;
  const int mt = tt / NT, nt = tt % NT;
  const __bf16* pb = P + ((long)bh * T + mt * 16) * T;
  const __bf16* vb = VT + ((long)bh * 784 + nt * 16) * 512;
  v8f acc = {};
  for (int kt = 0; kt < T / 32; ++kt) {
    v16bf a = load_frag_a(pb, T, kt * 32, lane);
    v16bf b = load_frag_b_rowK(vb, 512, kt * 32, lane);
    acc = wmma_bf16(a, b, acc);
  }
  const int n = lane & 15, hi = lane >> 4;
  float* op = O + ((long)bh * T + mt * 16) * 784 + nt * 16;
#pragma unroll
  for (int r = 0; r < 8; ++r) op[(long)(r + 8 * hi) * 784 + n] = acc[r];
}

// ---------------------------------------------------------------------------
// K8: output projection + LeakyReLU + LN over (C,F) per (b,t) + residual.
// ---------------------------------------------------------------------------
__global__ void k_proj_ln_res(const float* __restrict__ O, const float* __restrict__ Wp,
                              const float* __restrict__ bp, const float* __restrict__ ap,
                              const float* __restrict__ gp, const float* __restrict__ bgp,
                              const float* __restrict__ res, float* __restrict__ out,
                              int C, int T, int F, int H, int Dv) {
  const int bt = blockIdx.x;
  const int t = bt % T;
  const long b = bt / T;
  const int CF = C * F;
  const float a = ap[0];
  __shared__ float red[256];
  float vals[13];
  float lsum = 0.f;
#pragma unroll
  for (int ii = 0; ii < 13; ++ii) {
    int p = threadIdx.x + ii * 256;
    float v = 0.f;
    if (p < CF) {
      int co = p / F, f = p % F;
      float s = bp[co];
      for (int hd = 0; hd < C; ++hd) {       // Hh*Dv == C == 48
        int h = hd / Dv, dv = hd % Dv;
        float ov = O[((b * H + h) * T + t) * 784 + dv * F + f];
        s += ov * Wp[co * C + hd];
      }
      v = s >= 0.f ? s : a * s;
      lsum += v;
    }
    vals[ii] = v;
  }
  red[threadIdx.x] = lsum; __syncthreads();
  for (int o = 128; o > 0; o >>= 1) {
    if (threadIdx.x < o) red[threadIdx.x] += red[threadIdx.x + o];
    __syncthreads();
  }
  const float mu = red[0] / (float)CF; __syncthreads();
  float lvar = 0.f;
#pragma unroll
  for (int ii = 0; ii < 13; ++ii) {
    int p = threadIdx.x + ii * 256;
    if (p < CF) { float d = vals[ii] - mu; lvar += d * d; }
  }
  red[threadIdx.x] = lvar; __syncthreads();
  for (int o = 128; o > 0; o >>= 1) {
    if (threadIdx.x < o) red[threadIdx.x] += red[threadIdx.x + o];
    __syncthreads();
  }
  const float inv = rsqrtf(red[0] / (float)CF + 1e-5f);
#pragma unroll
  for (int ii = 0; ii < 13; ++ii) {
    int p = threadIdx.x + ii * 256;
    if (p < CF) {
      int co = p / F, f = p % F;
      long oi = ((b * C + co) * T + t) * F + f;
      out[oi] = (vals[ii] - mu) * inv * gp[co * F + f] + bgp[co * F + f] + res[oi];
    }
  }
}

// ---------------------------------------------------------------------------
// Host orchestration
// ---------------------------------------------------------------------------
extern "C" void kernel_launch(void* const* d_in, const int* in_sizes, int n_in,
                              void* d_out, int out_size, void* d_ws, size_t ws_size,
                              hipStream_t stream) {
  (void)in_sizes; (void)n_in; (void)out_size; (void)ws_size;
  const int B = 4, C = 48, T = 512, F = 65, Hh = 4, E = 8, Dv = 12;
  const long XN = (long)B * C * T * F;          // 6,389,760

  const float* x   = (const float*)d_in[0];
  const float* x2  = (const float*)d_in[1];
  const float* ng  = (const float*)d_in[2];
  const float* nb  = (const float*)d_in[3];
  const float* LWih = (const float*)d_in[4];    // (4,2,768,192)
  const float* LWhh = (const float*)d_in[5];    // (4,2,768,192)
  const float* Lb  = (const float*)d_in[6];     // (4,2,768)
  const float* TW  = (const float*)d_in[7];     // (4,384,48,4)
  const float* Tb  = (const float*)d_in[8];     // (4,48)
  const float* AWqk = (const float*)d_in[9];    // (2,2,4,8,48)
  const float* Abqk = (const float*)d_in[10];   // (2,2,4,8)
  const float* Aaqk = (const float*)d_in[11];   // (2,2,4)
  const float* Agqk = (const float*)d_in[12];   // (2,2,4,8,65)
  const float* Abgqk = (const float*)d_in[13];
  const float* AWv = (const float*)d_in[14];    // (2,4,12,48)
  const float* Abv = (const float*)d_in[15];    // (2,4,12)
  const float* Aav = (const float*)d_in[16];    // (2,4)
  const float* Agv = (const float*)d_in[17];    // (2,4,12,65)
  const float* Abgv = (const float*)d_in[18];
  const float* AWp = (const float*)d_in[19];    // (2,48,48)
  const float* Abp = (const float*)d_in[20];    // (2,48)
  const float* Aap = (const float*)d_in[21];    // (2,)
  const float* Agp = (const float*)d_in[22];    // (2,48,65)
  const float* Abgp = (const float*)d_in[23];

  // ------- workspace arena -------
  char* ws = (char*)d_ws;
  size_t off = 0;
  auto alloc = [&](size_t bytes) { char* p = ws + off; off += (bytes + 255) & ~(size_t)255; return p; };
  float*  z    = (float*)alloc(XN * 4);                         // LN output
  __bf16* U    = (__bf16*)alloc((size_t)272 * 509 * 192 * 2);   // unfolded (max)
  __bf16* WihS = (__bf16*)alloc((size_t)2 * 147456 * 2);
  __bf16* WhhS = (__bf16*)alloc((size_t)2 * 147456 * 2);
  float*  hs   = (float*)alloc((size_t)272 * 509 * 384 * 4);    // scan output (max)
  float*  A1   = (float*)alloc(XN * 4);
  float*  A2   = (float*)alloc(XN * 4);
  const int BH = B * Hh;
  __bf16* Qf = (__bf16*)alloc((size_t)BH * T * 544 * 2);
  __bf16* Kf = (__bf16*)alloc((size_t)BH * T * 544 * 2);
  __bf16* VT = (__bf16*)alloc((size_t)BH * 784 * 512 * 2);
  float*  S  = (float*)alloc((size_t)BH * T * T * 4);
  __bf16* P  = (__bf16*)alloc((size_t)BH * T * T * 2);
  float*  O  = (float*)alloc((size_t)BH * T * 784 * 4);

  const long TF = (long)T * F;
  const float scale = 1.0f / sqrtf((float)(E * F));

  for (int s = 0; s < 2; ++s) {
    const float* inp = (s == 0) ? x : x2;

    // ---------------- intra ----------------
    {
      const int st = s, L = F - 3, nSeq = B * T;                 // L = 62
      k_ln_c<<<(unsigned)((B * TF + 255) / 256), 256, 0, stream>>>(
          inp, ng + st * C, nb + st * C, z, C, TF, B * TF);
      long utot = (long)nSeq * L * 192;
      k_unfold<<<(unsigned)((utot + 255) / 256), 256, 0, stream>>>(
          z, U, C, T, F, L, utot, 0);
      for (int d = 0; d < 2; ++d) {
        k_pack_w<<<576, 256, 0, stream>>>(LWih + ((long)(st * 2 + d)) * 147456,
                                          WihS + (long)d * 147456);
        k_pack_w<<<576, 256, 0, stream>>>(LWhh + ((long)(st * 2 + d)) * 147456,
                                          WhhS + (long)d * 147456);
      }
      k_lstm_scan<<<dim3(nSeq / 16, 2), 384, WHH_BYTES + 2 * 16 * 192 * 2, stream>>>(
          U, WihS, WhhS, Lb + st * 1536, hs, L);
      long ctot = XN;
      k_convt_res<<<(unsigned)((ctot + 255) / 256), 256, 0, stream>>>(
          hs, TW + (long)st * 73728, Tb + st * C, inp, A1, C, T, F, L, ctot, 0);
    }

    // ---------------- inter ----------------
    {
      const int st = 2 + s, L = T - 3, nSeq = B * F, nSeqPad = 272; // L = 509
      k_ln_c<<<(unsigned)((B * TF + 255) / 256), 256, 0, stream>>>(
          A1, ng + st * C, nb + st * C, z, C, TF, B * TF);
      (void)hipMemsetAsync(U, 0, (size_t)nSeqPad * L * 192 * 2, stream);
      long utot = (long)nSeq * L * 192;
      k_unfold<<<(unsigned)((utot + 255) / 256), 256, 0, stream>>>(
          z, U, C, T, F, L, utot, 1);
      for (int d = 0; d < 2; ++d) {
        k_pack_w<<<576, 256, 0, stream>>>(LWih + ((long)(st * 2 + d)) * 147456,
                                          WihS + (long)d * 147456);
        k_pack_w<<<576, 256, 0, stream>>>(LWhh + ((long)(st * 2 + d)) * 147456,
                                          WhhS + (long)d * 147456);
      }
      k_lstm_scan<<<dim3(nSeqPad / 16, 2), 384, WHH_BYTES + 2 * 16 * 192 * 2, stream>>>(
          U, WihS, WhhS, Lb + st * 1536, hs, L);
      long ctot = XN;
      k_convt_res<<<(unsigned)((ctot + 255) / 256), 256, 0, stream>>>(
          hs, TW + (long)st * 73728, Tb + st * C, A1, A2, C, T, F, L, ctot, 1);
    }

    // ---------------- attention ----------------
    {
      (void)hipMemsetAsync(Qf, 0, (size_t)BH * T * 544 * 2, stream); // zero K padding
      (void)hipMemsetAsync(Kf, 0, (size_t)BH * T * 544 * 2, stream);
      (void)hipMemsetAsync(VT, 0, (size_t)BH * 784 * 512 * 2, stream); // zero N padding
      // Q / K heads
      for (int qk = 0; qk < 2; ++qk) {
        int pi = s * 2 + qk;
        k_heads<<<BH * T, 256, 0, stream>>>(
            A2, AWqk + (long)pi * Hh * E * C, Abqk + (long)pi * Hh * E,
            Aaqk + (long)pi * Hh, Agqk + (long)pi * Hh * E * F,
            Abgqk + (long)pi * Hh * E * F, (qk == 0) ? Qf : Kf,
            C, T, F, Hh, E, 544, 0);
      }
      // V head (transposed layout)
      k_heads<<<BH * T, 256, 0, stream>>>(
          A2, AWv + (long)s * Hh * Dv * C, Abv + (long)s * Hh * Dv,
          Aav + (long)s * Hh, Agv + (long)s * Hh * Dv * F,
          Abgv + (long)s * Hh * Dv * F, VT, C, T, F, Hh, Dv, 0, 1);
      // scores + softmax + O
      k_scores<<<(BH * 32 * 32) / 8, 256, 0, stream>>>(Qf, Kf, S, T, 17, 544, scale, 32);
      k_softmax<<<BH * T, 256, 0, stream>>>(S, P, T);
      k_attn_o<<<(BH * 32 * 49) / 8, 256, 0, stream>>>(P, VT, O, T, 49);
      // projection + LN + residual -> output slice
      k_proj_ln_res<<<B * T, 256, 0, stream>>>(
          O, AWp + (long)s * C * C, Abp + (long)s * C, Aap + s,
          Agp + (long)s * C * F, Abgp + (long)s * C * F, A2,
          (float*)d_out + (long)s * XN, C, T, F, Hh, Dv);
    }
  }
}
